// CareerTreeModel_75831942578729
// MI455X (gfx1250) — matrix operand order repaired and checked by reading, not compiled, not run on hardware
//
#include <hip/hip_runtime.h>
#include <hip/hip_bf16.h>
#include <cstdint>
#include <cstddef>

// ---------------- problem constants (match reference) ----------------
#define N_NODES 50000
#define N_EDGES 800000
#define N_PAIRS 800000
#define FDIM    128          // DIN = HEADS*HID = DOUT = 128
#define HEADS1  4
#define HID1    32
#define NEG_SLOPE 0.2f

typedef __attribute__((ext_vector_type(16))) __bf16 v16bf;
typedef __attribute__((ext_vector_type(8)))  __bf16 v8bf;
typedef __attribute__((ext_vector_type(8)))  float  v8f;

union AFrag { v16bf v; v8bf h[2]; };

static __device__ __forceinline__ v8f wmma_bf16(v16bf a, v16bf b, v8f c) {
  // (neg_a, A, neg_b, B, c_mod, C, reuse_a, reuse_b)
  return __builtin_amdgcn_wmma_f32_16x16x32_bf16(false, a, false, b, (short)0, c,
                                                 false, false);
}

// order-preserving float<->uint encoding for atomicMax-based segment max
static __device__ __forceinline__ unsigned f2ord(float f) {
  unsigned u = __float_as_uint(f);
  return (u & 0x80000000u) ? ~u : (u | 0x80000000u);
}
static __device__ __forceinline__ float ord2f(unsigned u) {
  return (u & 0x80000000u) ? __uint_as_float(u & 0x7fffffffu)
                           : __uint_as_float(~u);
}

static __device__ __forceinline__ void edge_sd(const int* __restrict__ ei,
                                               long long e, int& s, int& d) {
  if (e < (long long)N_EDGES) { s = ei[e]; d = ei[N_EDGES + e]; }
  else { s = d = (int)(e - N_EDGES); }   // self loops appended
}

// ---------------------------------------------------------------------
// Pack f32 weight W[K x Nc] (row-major) into WMMA B-operand lane layout:
// per (ntile, ktile): 32 lanes x 16 contiguous bf16.
// lane<16  -> column ntile*16+lane, K = ktile*32 + {0..7, 16..23}
// lane>=16 -> column ntile*16+lane-16, K = ktile*32 + {8..15, 24..31}
// ---------------------------------------------------------------------
__global__ void pack_b_kernel(const float* __restrict__ W,
                              unsigned short* __restrict__ Bp,
                              int K, int Nc) {
  int tid = blockIdx.x * blockDim.x + threadIdx.x;
  int numKt = K >> 5, numNt = Nc >> 4;
  int total = numKt * numNt * 32;
  if (tid >= total) return;
  int lane = tid & 31;
  int t    = tid >> 5;
  int kt   = t % numKt;
  int nt   = t / numKt;
  int col  = nt * 16 + (lane & 15);
  int kb   = kt * 32 + (lane >> 4) * 8;
  __bf16* dst = reinterpret_cast<__bf16*>(Bp) +
                ((size_t)(nt * numKt + kt) * 32 + lane) * 16;
#pragma unroll
  for (int j = 0; j < 16; ++j)
    dst[j] = (__bf16)W[(size_t)(kb + (j & 7) + (j >> 3) * 16) * Nc + col];
}

// ---------------- activation precision conversion --------------------
__global__ void f32_to_bf16_kernel(const float* __restrict__ in,
                                   unsigned short* __restrict__ out, size_t n) {
  size_t t = (size_t)blockIdx.x * blockDim.x + threadIdx.x;
  if (t < n) reinterpret_cast<__bf16*>(out)[t] = (__bf16)in[t];
}

// ---------------------------------------------------------------------
// C[M x 128] = A[M x 128] (bf16, row-major) @ Bpacked (bf16). One wave per
// 16x16 tile, 8 waves/block cover a 128-wide stripe of 16 rows. M%16==0.
// A-fragment = two aligned 16B loads per k-tile, zero conversion VALU.
// ---------------------------------------------------------------------
__global__ __launch_bounds__(256)
void gemm128_bf16_kernel(const unsigned short* __restrict__ A,
                         const unsigned short* __restrict__ Bp,
                         float* __restrict__ out) {
  const int lane  = threadIdx.x & 31;
  const int ntile = threadIdx.x >> 5;       // 0..7
  const int half  = lane >> 4;              // 0 or 1
  const int l16   = lane & 15;
  const int row   = blockIdx.x * 16 + l16;
  const __bf16* __restrict__ arow =
      reinterpret_cast<const __bf16*>(A) + (size_t)row * FDIM;

  v8f acc = {0.f, 0.f, 0.f, 0.f, 0.f, 0.f, 0.f, 0.f};
#pragma unroll
  for (int kt = 0; kt < 4; ++kt) {          // K = 128 = 4 * 32
    const int kb = kt * 32 + half * 8;
    AFrag a;
    a.h[0] = *reinterpret_cast<const v8bf*>(arow + kb);
    a.h[1] = *reinterpret_cast<const v8bf*>(arow + kb + 16);
    v16bf b = *reinterpret_cast<const v16bf*>(
        Bp + ((size_t)(ntile * 4 + kt) * 32 + lane) * 16);
    acc = wmma_bf16(a.v, b, acc);
  }
  // D layout: vgpr r -> row (half*8 + r), col = l16 within tile
#pragma unroll
  for (int r = 0; r < 8; ++r) {
    int orow = blockIdx.x * 16 + half * 8 + r;
    out[(size_t)orow * FDIM + ntile * 16 + l16] = acc[r];
  }
}

// ---------------- attention coefficient kernels ----------------------
__global__ void alpha_kernel(const float* __restrict__ xw,
                             const float* __restrict__ a_src,
                             const float* __restrict__ a_dst,
                             float* __restrict__ as, float* __restrict__ ad,
                             int n, int heads, int C) {
  int t = blockIdx.x * blockDim.x + threadIdx.x;
  if (t >= n * heads) return;
  int node = t / heads, h = t % heads;
  const float* v = xw + (size_t)node * (heads * C) + h * C;
  float ss = 0.f, sd = 0.f;
  for (int c = 0; c < C; ++c) {
    float x = v[c];
    ss += x * a_src[h * C + c];
    sd += x * a_dst[h * C + c];
  }
  as[t] = ss;
  ad[t] = sd;
}

__global__ void fill_u32_kernel(unsigned* __restrict__ p, unsigned v, size_t n) {
  size_t t = (size_t)blockIdx.x * blockDim.x + threadIdx.x;
  if (t < n) p[t] = v;
}

__global__ void logit_max_kernel(const int* __restrict__ ei,
                                 const float* __restrict__ as,
                                 const float* __restrict__ ad,
                                 float* __restrict__ logits,
                                 unsigned* __restrict__ amax, int heads) {
  long long t = (long long)blockIdx.x * blockDim.x + threadIdx.x;
  long long total = (long long)(N_EDGES + N_NODES) * heads;
  if (t >= total) return;
  long long e = t / heads;
  int h = (int)(t % heads);
  int s, d;
  edge_sd(ei, e, s, d);
  float l = as[(size_t)s * heads + h] + ad[(size_t)d * heads + h];
  l = (l > 0.f) ? l : NEG_SLOPE * l;
  logits[t] = l;
  atomicMax(&amax[(size_t)d * heads + h], f2ord(l));
}

__global__ void exp_den_kernel(const int* __restrict__ ei,
                               float* __restrict__ logits,
                               const unsigned* __restrict__ amax,
                               float* __restrict__ den, int heads) {
  long long t = (long long)blockIdx.x * blockDim.x + threadIdx.x;
  long long total = (long long)(N_EDGES + N_NODES) * heads;
  if (t >= total) return;
  long long e = t / heads;
  int h = (int)(t % heads);
  int s, d;
  edge_sd(ei, e, s, d);
  float m  = ord2f(amax[(size_t)d * heads + h]);
  float ex = __expf(logits[t] - m);
  logits[t] = ex;
  atomicAdd(&den[(size_t)d * heads + h], ex);
}

__global__ void scatter_kernel(const int* __restrict__ ei,
                               const float* __restrict__ xw,
                               const float* __restrict__ ex,
                               const float* __restrict__ den,
                               float* __restrict__ outacc, int heads, int C) {
  long long t = (long long)blockIdx.x * blockDim.x + threadIdx.x;
  long long total = (long long)(N_EDGES + N_NODES) * FDIM;
  if (t >= total) return;
  int f = (int)(t % FDIM);
  long long e = t / FDIM;
  int h = f / C;
  int s, d;
  edge_sd(ei, e, s, d);
  float w = ex[(size_t)e * heads + h] / den[(size_t)d * heads + h];
  atomicAdd(&outacc[(size_t)d * FDIM + f], xw[(size_t)s * FDIM + f] * w);
}

// bias (+ optional ELU) epilogue, emitting bf16 activations for the next GEMM
__global__ void bias_act_bf16_kernel(const float* __restrict__ buf,
                                     const float* __restrict__ bias,
                                     unsigned short* __restrict__ outbf,
                                     size_t n, int act) {
  size_t t = (size_t)blockIdx.x * blockDim.x + threadIdx.x;
  if (t >= n) return;
  float v = buf[t] + bias[t % FDIM];
  if (act) v = (v > 0.f) ? v : (__expf(v) - 1.f);   // ELU(alpha=1)
  reinterpret_cast<__bf16*>(outbf)[t] = (__bf16)v;
}

// ---------------------------------------------------------------------
// Fused edge MLP: per wave, 16 pairs.
//   e1 = relu([z_i|z_j] @ mw1 + mb1)   16x256 @ 256x128  (64 wmma)
//   e2 = relu(e1 @ mw2 + mb2)          16x128 @ 128x64   (16 wmma)
//   out = sigmoid(e2 @ mw3 + mb3)      64-dot per row
// z pre-converted to bf16: A-fragments are pure b128 gathers.
// LDS staging per wave; same-wave DS ops are in-order on CDNA5.
// ---------------------------------------------------------------------
#define MLP_WAVES 4
__global__ __launch_bounds__(128)
void edge_mlp_kernel(const unsigned short* __restrict__ zb,
                     const int* __restrict__ pairs,
                     const unsigned short* __restrict__ B1p,
                     const float* __restrict__ mb1,
                     const unsigned short* __restrict__ B2p,
                     const float* __restrict__ mb2,
                     const float* __restrict__ mw3,
                     const float* __restrict__ mb3,
                     float* __restrict__ out) {
  __shared__ float lds[MLP_WAVES][16][132];
  const int lane = threadIdx.x & 31;
  const int wave = threadIdx.x >> 5;
  const int tile = blockIdx.x * MLP_WAVES + wave;
  const int half = lane >> 4;
  const int l16  = lane & 15;
  const int p    = tile * 16 + l16;
  const int ii   = pairs[p];
  const int jj   = pairs[N_PAIRS + p];
  const __bf16* __restrict__ zi =
      reinterpret_cast<const __bf16*>(zb) + (size_t)ii * FDIM;
  const __bf16* __restrict__ zj =
      reinterpret_cast<const __bf16*>(zb) + (size_t)jj * FDIM;
  float (*S)[132] = lds[wave];

  // ---- stage 1: 16x256 @ 256x128 ----
  v8f acc[8];
#pragma unroll
  for (int nt = 0; nt < 8; ++nt)
    acc[nt] = (v8f){0.f, 0.f, 0.f, 0.f, 0.f, 0.f, 0.f, 0.f};
#pragma unroll
  for (int kt = 0; kt < 8; ++kt) {          // K = 256
    const int kb = kt * 32 + half * 8;
    const int k0 = kb, k1 = kb + 16;        // 8-elem blocks never straddle 128
    const __bf16* s0 = (k0 < FDIM) ? (zi + k0) : (zj + k0 - FDIM);
    const __bf16* s1 = (k1 < FDIM) ? (zi + k1) : (zj + k1 - FDIM);
    AFrag a;
    a.h[0] = *reinterpret_cast<const v8bf*>(s0);
    a.h[1] = *reinterpret_cast<const v8bf*>(s1);
#pragma unroll
    for (int nt = 0; nt < 8; ++nt) {
      v16bf b = *reinterpret_cast<const v16bf*>(
          B1p + ((size_t)(nt * 8 + kt) * 32 + lane) * 16);
      acc[nt] = wmma_bf16(a.v, b, acc[nt]);
    }
  }
#pragma unroll
  for (int nt = 0; nt < 8; ++nt) {
    float bias = mb1[nt * 16 + l16];
#pragma unroll
    for (int r = 0; r < 8; ++r) {
      float v = acc[nt][r] + bias;
      S[half * 8 + r][nt * 16 + l16] = (v > 0.f) ? v : 0.f;
    }
  }

  // ---- stage 2: 16x128 @ 128x64 ----
  v8f acc2[4];
#pragma unroll
  for (int nt = 0; nt < 4; ++nt)
    acc2[nt] = (v8f){0.f, 0.f, 0.f, 0.f, 0.f, 0.f, 0.f, 0.f};
#pragma unroll
  for (int kt = 0; kt < 4; ++kt) {          // K = 128
    const int kb = kt * 32 + half * 8;
    v16bf a;
#pragma unroll
    for (int t = 0; t < 8; ++t) {
      a[t]     = (__bf16)S[l16][kb + t];
      a[8 + t] = (__bf16)S[l16][kb + 16 + t];
    }
#pragma unroll
    for (int nt = 0; nt < 4; ++nt) {
      v16bf b = *reinterpret_cast<const v16bf*>(
          B2p + ((size_t)(nt * 4 + kt) * 32 + lane) * 16);
      acc2[nt] = wmma_bf16(a, b, acc2[nt]);
    }
  }
#pragma unroll
  for (int nt = 0; nt < 4; ++nt) {
    float bias = mb2[nt * 16 + l16];
#pragma unroll
    for (int r = 0; r < 8; ++r) {
      float v = acc2[nt][r] + bias;
      S[half * 8 + r][nt * 16 + l16] = (v > 0.f) ? v : 0.f;
    }
  }

  // ---- stage 3: 64-dot + sigmoid ----
  if (lane < 16) {
    float s = mb3[0];
#pragma unroll 8
    for (int c = 0; c < 64; ++c) s += S[lane][c] * mw3[c];
    out[tile * 16 + lane] = 1.f / (1.f + __expf(-s));
  }
}

// ---------------------------------------------------------------------
extern "C" void kernel_launch(void* const* d_in, const int* in_sizes, int n_in,
                              void* d_out, int out_size, void* d_ws, size_t ws_size,
                              hipStream_t stream) {
  const float* x      = (const float*)d_in[0];
  const int*   ei     = (const int*)d_in[1];
  const int*   pairs  = (const int*)d_in[2];
  const float* W1     = (const float*)d_in[3];
  const float* a_src1 = (const float*)d_in[4];
  const float* a_dst1 = (const float*)d_in[5];
  const float* b1     = (const float*)d_in[6];
  const float* W2     = (const float*)d_in[7];
  const float* a_src2 = (const float*)d_in[8];
  const float* a_dst2 = (const float*)d_in[9];
  const float* b2     = (const float*)d_in[10];
  const float* mw1    = (const float*)d_in[11];
  const float* mb1    = (const float*)d_in[12];
  const float* mw2    = (const float*)d_in[13];
  const float* mb2    = (const float*)d_in[14];
  const float* mw3    = (const float*)d_in[15];
  const float* mb3    = (const float*)d_in[16];
  float* out = (float*)d_out;

  // workspace carve-up
  char* base = (char*)d_ws;
  size_t off = 0;
  auto alloc = [&](size_t bytes) -> void* {
    void* p = base + off;
    off = (off + bytes + 255) & ~(size_t)255;
    return p;
  };
  unsigned short* W1p = (unsigned short*)alloc((size_t)128 * 128 * 2);
  unsigned short* W2p = (unsigned short*)alloc((size_t)128 * 128 * 2);
  unsigned short* M1p = (unsigned short*)alloc((size_t)256 * 128 * 2);
  unsigned short* M2p = (unsigned short*)alloc((size_t)128 * 64 * 2);
  float* bufA  = (float*)alloc((size_t)N_NODES * FDIM * 4);   // xw (reused layer 2)
  float* bufH  = (float*)alloc((size_t)N_NODES * FDIM * 4);   // h accumulator
  float* bufZ  = (float*)alloc((size_t)N_NODES * FDIM * 4);   // z accumulator
  unsigned short* abf = (unsigned short*)alloc((size_t)N_NODES * FDIM * 2);
  // ^ bf16 activation buffer, reused serially for x -> h -> z
  float* as1   = (float*)alloc((size_t)N_NODES * HEADS1 * 4);
  float* ad1   = (float*)alloc((size_t)N_NODES * HEADS1 * 4);
  unsigned* am1 = (unsigned*)alloc((size_t)N_NODES * HEADS1 * 4);
  float* den1  = (float*)alloc((size_t)N_NODES * HEADS1 * 4);
  float* ex1   = (float*)alloc((size_t)(N_EDGES + N_NODES) * HEADS1 * 4);
  float* as2   = (float*)alloc((size_t)N_NODES * 4);
  float* ad2   = (float*)alloc((size_t)N_NODES * 4);
  unsigned* am2 = (unsigned*)alloc((size_t)N_NODES * 4);
  float* den2  = (float*)alloc((size_t)N_NODES * 4);
  float* ex2   = (float*)alloc((size_t)(N_EDGES + N_NODES) * 4);

  const unsigned ORD_NEG_INF = 0x007FFFFFu;   // f2ord(-inf)
  auto cdiv = [](long long a, long long b) { return (unsigned)((a + b - 1) / b); };
  const size_t NF = (size_t)N_NODES * FDIM;

  // ---- pack weights to bf16 WMMA B layout ----
  pack_b_kernel<<<cdiv(8 * 4 * 32, 256), 256, 0, stream>>>(W1,  W1p, 128, 128);
  pack_b_kernel<<<cdiv(8 * 4 * 32, 256), 256, 0, stream>>>(W2,  W2p, 128, 128);
  pack_b_kernel<<<cdiv(8 * 8 * 32, 256), 256, 0, stream>>>(mw1, M1p, 256, 128);
  pack_b_kernel<<<cdiv(4 * 4 * 32, 256), 256, 0, stream>>>(mw2, M2p, 128, 64);

  // ================= GAT layer 1 (heads=4, C=32) =================
  f32_to_bf16_kernel<<<cdiv(NF, 256), 256, 0, stream>>>(x, abf, NF);
  gemm128_bf16_kernel<<<N_NODES / 16, 256, 0, stream>>>(abf, W1p, bufA);
  alpha_kernel<<<cdiv((long long)N_NODES * HEADS1, 256), 256, 0, stream>>>(
      bufA, a_src1, a_dst1, as1, ad1, N_NODES, HEADS1, HID1);
  fill_u32_kernel<<<cdiv((long long)N_NODES * HEADS1, 256), 256, 0, stream>>>(
      am1, ORD_NEG_INF, (size_t)N_NODES * HEADS1);
  fill_u32_kernel<<<cdiv((long long)N_NODES * HEADS1, 256), 256, 0, stream>>>(
      (unsigned*)den1, 0u, (size_t)N_NODES * HEADS1);
  fill_u32_kernel<<<cdiv(NF, 256), 256, 0, stream>>>(
      (unsigned*)bufH, 0u, NF);
  logit_max_kernel<<<cdiv((long long)(N_EDGES + N_NODES) * HEADS1, 256), 256, 0,
                     stream>>>(ei, as1, ad1, ex1, am1, HEADS1);
  exp_den_kernel<<<cdiv((long long)(N_EDGES + N_NODES) * HEADS1, 256), 256, 0,
                   stream>>>(ei, ex1, am1, den1, HEADS1);
  scatter_kernel<<<cdiv((long long)(N_EDGES + N_NODES) * FDIM, 256), 256, 0,
                   stream>>>(ei, bufA, ex1, den1, bufH, HEADS1, HID1);
  bias_act_bf16_kernel<<<cdiv(NF, 256), 256, 0, stream>>>(
      bufH, b1, abf, NF, 1);                  // + bias, ELU -> bf16 h

  // ================= GAT layer 2 (heads=1, C=128) =================
  gemm128_bf16_kernel<<<N_NODES / 16, 256, 0, stream>>>(abf, W2p, bufA);
  alpha_kernel<<<cdiv((long long)N_NODES, 256), 256, 0, stream>>>(
      bufA, a_src2, a_dst2, as2, ad2, N_NODES, 1, FDIM);
  fill_u32_kernel<<<cdiv((long long)N_NODES, 256), 256, 0, stream>>>(
      am2, ORD_NEG_INF, (size_t)N_NODES);
  fill_u32_kernel<<<cdiv((long long)N_NODES, 256), 256, 0, stream>>>(
      (unsigned*)den2, 0u, (size_t)N_NODES);
  fill_u32_kernel<<<cdiv(NF, 256), 256, 0, stream>>>(
      (unsigned*)bufZ, 0u, NF);
  logit_max_kernel<<<cdiv((long long)(N_EDGES + N_NODES), 256), 256, 0, stream>>>(
      ei, as2, ad2, ex2, am2, 1);
  exp_den_kernel<<<cdiv((long long)(N_EDGES + N_NODES), 256), 256, 0, stream>>>(
      ei, ex2, am2, den2, 1);
  scatter_kernel<<<cdiv((long long)(N_EDGES + N_NODES) * FDIM, 256), 256, 0,
                   stream>>>(ei, bufA, ex2, den2, bufZ, 1, FDIM);
  bias_act_bf16_kernel<<<cdiv(NF, 256), 256, 0, stream>>>(
      bufZ, b2, abf, NF, 0);                  // + bias only -> bf16 z

  // ================= fused edge MLP head =================
  edge_mlp_kernel<<<N_PAIRS / (16 * MLP_WAVES), 128, 0, stream>>>(
      abf, pairs, M1p, mb1, M2p, mb2, mw3, mb3, out);
}
